// RGAT_69672959475784
// MI455X (gfx1250) — matrix operand order repaired and checked by reading, not compiled, not run on hardware
//
#include <hip/hip_runtime.h>

// ---------------------------------------------------------------------------
// R-GAT forward for MI455X (gfx1250, wave32). Heavy GEMMs go through
// v_wmma_f32_16x16x32_bf16 (f32 accumulate) with async-to-LDS A staging
// (global_load_async_to_lds_b128, ASYNCcnt). Scatter/softmax/BN are
// bandwidth-bound elementwise kernels.
// ---------------------------------------------------------------------------

typedef unsigned int  u32;
typedef unsigned short u16;
typedef __bf16 bf16_t;
typedef bf16_t v16bf __attribute__((ext_vector_type(16)));
typedef float  v8f   __attribute__((ext_vector_type(8)));
typedef u32    v4u   __attribute__((ext_vector_type(4)));

constexpr int CC    = 768;
constexpr int HIDN  = 512;
constexpr int HEADS = 4;
constexpr int DHD   = 128;
constexpr int NE    = 3;
constexpr int NSRC  = 100000;
constexpr int NDST  = 25000;
constexpr int EDGES = 150000;
constexpr int LBL   = 10000;
constexpr int NCLS  = 153;
constexpr int NCLSP = 256;   // mlp_W2 column padding so B loads are unpredicated

__device__ inline u16 f2bf(float f) {
  u32 b = __float_as_uint(f);
  u32 r = b + 0x7FFFu + ((b >> 16) & 1u);   // round-to-nearest-even
  return (u16)(r >> 16);
}

__device__ inline void atomicAddF(float* p, float v) {
  __hip_atomic_fetch_add(p, v, __ATOMIC_RELAXED, __HIP_MEMORY_SCOPE_AGENT);
}

__device__ inline void atomicMaxF(float* p, float v) {
  if (v >= 0.f) atomicMax((int*)p, __float_as_int(v));
  else          atomicMin((unsigned int*)p, (unsigned int)__float_as_int(v));
}

// CDNA5 async DMA: copy 16 bytes global -> LDS, tracked by ASYNCcnt.
__device__ inline void async_ld16(u32 lds_byte_addr, const u16* g) {
  asm volatile("global_load_async_to_lds_b128 %0, %1, off"
               :: "v"(lds_byte_addr), "v"(g) : "memory");
}
__device__ inline void wait_async0() {
  asm volatile("s_wait_asynccnt 0x0" ::: "memory");
}

// --------------------------- elementwise helpers ---------------------------

__global__ void k_f32_to_bf16(const float* __restrict__ s, u16* __restrict__ d, long n) {
  long i = (long)blockIdx.x * blockDim.x + threadIdx.x;
  if (i < n) d[i] = f2bf(s[i]);
}

// convert with column zero-padding: d is K x ldp, s is K x N
__global__ void k_cvt_pad(const float* __restrict__ s, u16* __restrict__ d, int K, int N, int ldp) {
  int i = blockIdx.x * blockDim.x + threadIdx.x;
  if (i >= K * ldp) return;
  int k = i / ldp, n = i % ldp;
  d[i] = (n < N) ? f2bf(s[(long)k * N + n]) : (u16)0;
}

__global__ void k_fill(float* __restrict__ p, float v, long n) {
  long i = (long)blockIdx.x * blockDim.x + threadIdx.x;
  if (i < n) p[i] = v;
}

// per-column mean / rstd over n rows (biased variance, eps=1e-5)
__global__ __launch_bounds__(256) void k_colstats(const float* __restrict__ X, int n, int c,
                                                  float* __restrict__ mean, float* __restrict__ rstd) {
  int col = blockIdx.x;
  float s = 0.f, s2 = 0.f;
  for (int i = threadIdx.x; i < n; i += 256) {
    float v = X[(long)i * c + col];
    s += v; s2 += v * v;
  }
  __shared__ float sh0[256], sh1[256];
  sh0[threadIdx.x] = s; sh1[threadIdx.x] = s2; __syncthreads();
  for (int o = 128; o > 0; o >>= 1) {
    if (threadIdx.x < (unsigned)o) { sh0[threadIdx.x] += sh0[threadIdx.x + o]; sh1[threadIdx.x] += sh1[threadIdx.x + o]; }
    __syncthreads();
  }
  if (threadIdx.x == 0) {
    float mu  = sh0[0] / n;
    float var = sh1[0] / n - mu * mu;
    mean[col] = mu;
    rstd[col] = rsqrtf(var + 1e-5f);
  }
}

// act: 0=none, 1=relu, 2=elu
__global__ void k_bn_act(const float* __restrict__ X, const float* __restrict__ g, const float* __restrict__ be,
                         const float* __restrict__ mean, const float* __restrict__ rstd,
                         float* __restrict__ outf, u16* __restrict__ outbf, long total, int c, int act) {
  long i = (long)blockIdx.x * blockDim.x + threadIdx.x;
  if (i >= total) return;
  int col = (int)(i % c);
  float v = (X[i] - mean[col]) * rstd[col] * g[col] + be[col];
  if (act == 1)      v = fmaxf(v, 0.f);
  else if (act == 2) v = v > 0.f ? v : (__expf(v) - 1.f);
  if (outf)  outf[i]  = v;
  if (outbf) outbf[i] = f2bf(v);
}

// Lin[r, 0:768] = bf16(label_emb[label_y[r]]) ; Lin[r, 768:1536] = x1[label_idx[r]]
__global__ void k_label_gather(const float* __restrict__ emb, const int* __restrict__ ly,
                               const u16* __restrict__ x1, const int* __restrict__ lidx,
                               u16* __restrict__ Lin, long total) {
  long i = (long)blockIdx.x * blockDim.x + threadIdx.x;
  if (i >= total) return;
  int r = (int)(i / (2 * CC)), col = (int)(i % (2 * CC));
  if (col < CC) Lin[i] = f2bf(emb[(long)ly[r] * CC + col]);
  else          Lin[i] = x1[(long)lidx[r] * CC + (col - CC)];
}

__global__ void k_label_scatter(const float* __restrict__ h2, const int* __restrict__ lidx,
                                u16* __restrict__ x1, long total) {
  long i = (long)blockIdx.x * blockDim.x + threadIdx.x;
  if (i >= total) return;
  int r = (int)(i / CC), col = (int)(i % CC);
  x1[(long)lidx[r] * CC + col] = f2bf(h2[i]);
}

// el[i,h] = sum_k fs[i, h*DH+k] * attn[h,k]   (float4 / b128 loads)
__global__ void k_dot_heads(const float* __restrict__ fs, const float* __restrict__ attn,
                            float* __restrict__ out, int n) {
  int i = blockIdx.x * blockDim.x + threadIdx.x;
  if (i >= n * HEADS) return;
  int r = i >> 2, h = i & 3;
  const float4* f4 = (const float4*)(fs + (long)r * HIDN + h * DHD);
  const float4* a4 = (const float4*)(attn + h * DHD);
  float s = 0.f;
  #pragma unroll 4
  for (int k = 0; k < DHD / 4; k++) {
    float4 fv = f4[k], av = a4[k];
    s += fv.x * av.x + fv.y * av.y + fv.z * av.z + fv.w * av.w;
  }
  out[i] = s;
}

__global__ void k_edge_max(const int* __restrict__ src, const int* __restrict__ dst,
                           const float* __restrict__ el, const float* __restrict__ er,
                           float* __restrict__ ebuf, float* __restrict__ m, int E) {
  int i = blockIdx.x * blockDim.x + threadIdx.x;
  if (i >= E * HEADS) return;
  int e = i >> 2, h = i & 3;
  float v = el[src[e] * HEADS + h] + er[dst[e] * HEADS + h];
  v = v > 0.f ? v : 0.2f * v;                 // leaky_relu(0.2)
  ebuf[i] = v;
  atomicMaxF(&m[dst[e] * HEADS + h], v);
}

__global__ void k_fix_nonfinite(float* __restrict__ p, long n) {
  long i = (long)blockIdx.x * blockDim.x + threadIdx.x;
  if (i >= n) return;
  float v = p[i];
  if (!(v >= -3.4e38f && v <= 3.4e38f)) p[i] = 0.f;
}

__global__ void k_edge_exp(const int* __restrict__ dst, float* __restrict__ ebuf,
                           const float* __restrict__ m, float* __restrict__ den, int E) {
  int i = blockIdx.x * blockDim.x + threadIdx.x;
  if (i >= E * HEADS) return;
  int e = i >> 2, h = i & 3;
  float ex = __expf(ebuf[i] - m[dst[e] * HEADS + h]);
  ebuf[i] = ex;
  atomicAddF(&den[dst[e] * HEADS + h], ex);
}

// one block (128 threads) per edge; scatter-add 512 values
__global__ __launch_bounds__(DHD) void k_edge_aggr(const int* __restrict__ src, const int* __restrict__ dst,
                                                   const float* __restrict__ ebuf, const float* __restrict__ den,
                                                   const float* __restrict__ fs, float* __restrict__ rst, int E) {
  int e = blockIdx.x;
  if (e >= E) return;
  int t = threadIdx.x;
  int s = src[e], d = dst[e];
  #pragma unroll
  for (int h = 0; h < HEADS; h++) {
    float a = ebuf[(long)e * HEADS + h] / fmaxf(den[d * HEADS + h], 1e-9f);
    atomicAddF(&rst[(long)d * HIDN + h * DHD + t], a * fs[(long)s * HIDN + h * DHD + t]);
  }
}

// attention pooling over {conv0, conv1, conv2, sk}
__global__ __launch_bounds__(DHD) void k_pool(const float* __restrict__ hconv, const float* __restrict__ sk,
                                              const float* __restrict__ paW, const float* __restrict__ pab,
                                              float* __restrict__ fused, int ndst) {
  int row = blockIdx.x, t = threadIdx.x;
  __shared__ float red[DHD];
  __shared__ float w[4];
  const float* xs[4] = { hconv + (long)row * HIDN,
                         hconv + (long)ndst * HIDN + (long)row * HIDN,
                         hconv + 2L * ndst * HIDN + (long)row * HIDN,
                         sk + (long)row * HIDN };
  for (int p = 0; p < 4; p++) {
    float s = 0.f;
    for (int k = t; k < HIDN; k += DHD) s += xs[p][k] * paW[k];
    red[t] = s; __syncthreads();
    for (int o = DHD / 2; o > 0; o >>= 1) { if (t < o) red[t] += red[t + o]; __syncthreads(); }
    if (t == 0) w[p] = red[0] + pab[0];
    __syncthreads();
  }
  if (t == 0) {
    float mx = fmaxf(fmaxf(w[0], w[1]), fmaxf(w[2], w[3]));
    float e0 = __expf(w[0] - mx), e1 = __expf(w[1] - mx), e2 = __expf(w[2] - mx), e3 = __expf(w[3] - mx);
    float s = e0 + e1 + e2 + e3;
    w[0] = e0 / s; w[1] = e1 / s; w[2] = e2 / s; w[3] = e3 / s;
  }
  __syncthreads();
  for (int k = t; k < HIDN; k += DHD)
    fused[(long)row * HIDN + k] = w[0] * xs[0][k] + w[1] * xs[1][k] + w[2] * xs[2][k] + w[3] * xs[3][k];
}

__global__ __launch_bounds__(256) void k_softmax(const float* __restrict__ X, float* __restrict__ out, int n, int c) {
  int row = blockIdx.x, t = threadIdx.x;
  __shared__ float sh[256];
  float mx = -3.4e38f;
  for (int j = t; j < c; j += 256) mx = fmaxf(mx, X[(long)row * c + j]);
  sh[t] = mx; __syncthreads();
  for (int o = 128; o > 0; o >>= 1) { if (t < o) sh[t] = fmaxf(sh[t], sh[t + o]); __syncthreads(); }
  mx = sh[0]; __syncthreads();
  float s = 0.f;
  for (int j = t; j < c; j += 256) s += __expf(X[(long)row * c + j] - mx);
  sh[t] = s; __syncthreads();
  for (int o = 128; o > 0; o >>= 1) { if (t < o) sh[t] += sh[t + o]; __syncthreads(); }
  s = sh[0]; __syncthreads();
  for (int j = t; j < c; j += 256) out[(long)row * c + j] = __expf(X[(long)row * c + j] - mx) / s;
}

// ------------------------------- WMMA GEMM ---------------------------------
// C[M,N] = A[M,K] @ B[K,N] (+bias[N]) (+residual[M,N]); A,B bf16, acc f32.
// Block tile 128x128, BK=32; 8 wave32 waves, each 32x64 = 2x4 WMMA frags.
//
// A tile staged via global_load_async_to_lds_b128 into 64B rows with XOR
// chunk swizzle (16B-aligned chunks -> A fragments load as two ds_load_b128).
// M-edge handled by clamping the row address: garbage A rows only feed D
// rows the predicated epilogue never stores.
// B tile staged with full u32 loads (K%32==0 always; columns padded so
// n0+BN <= ldb always) and repacked to K-pair u32s via ds_store_b16.
// All global pointers are hoisted out of the K loop and advanced by strides.

constexpr int BM = 128, BN = 128, BK = 32;
constexpr int BS_STRIDE = BN + 2;              // u32 stride of packed-B rows
constexpr int AS_WORDS  = BM * 16;             // 2048 u32 (64B per row)
constexpr int BS_WORDS  = (BK / 2) * BS_STRIDE;

__global__ __launch_bounds__(256) void k_gemm_bf16(
    const u16* __restrict__ A, const u16* __restrict__ B,
    const float* __restrict__ bias, const float* __restrict__ residual,
    float* __restrict__ Cf, u16* __restrict__ Cbf,
    int M, int N, int K, int lda, int ldb, int ldc)
{
  __shared__ __align__(16) u32 smem[AS_WORDS + BS_WORDS];
  u32* As   = smem;
  u32* Bs   = smem + AS_WORDS;
  u16* Bs16 = (u16*)Bs;

  const int m0 = blockIdx.y * BM, n0 = blockIdx.x * BN;
  const int tid = threadIdx.x;
  const int wid = tid >> 5, lane = tid & 31;
  const int wrow = (wid & 3) * 32, wcol = (wid >> 2) * 64;
  const int lhalf = lane >> 4, lm = lane & 15;

  // low 32 bits of the flat pointer == wave-relative LDS byte address
  const u32 ldsA = (u32)(size_t)(&smem[0]);

  // ---- loop-invariant A async staging state (2 chunks / thread) ----
  const u16* ga[2];
  u32 ldsa[2];
  #pragma unroll
  for (int q = 0; q < 2; q++) {
    int c = tid * 2 + q;
    int m = c >> 2, t = c & 3;                  // 4 chunks per 64B row
    int gm = m0 + m; if (gm > M - 1) gm = M - 1;
    ga[q]   = A + (size_t)gm * lda + t * 8;
    ldsa[q] = ldsA + (u32)(m * 64 + ((t ^ ((m >> 2) & 3)) * 16));
  }
  // ---- loop-invariant B staging state (fixed column, rows kbase+4t) ----
  const int kbase = tid >> 6;                   // 0..3
  const int nn    = (tid & 63) * 2;
  const u16* gb   = B + (size_t)kbase * ldb + (n0 + nn);
  const int  kb   = kbase & 1;
  const int  kpb  = kbase >> 1;
  const int  bs0  = (kpb * BS_STRIDE + nn) * 2 + kb;   // u16 index, += 4*BS_STRIDE per t

  v8f acc[2][4];
  #pragma unroll
  for (int i = 0; i < 2; i++)
    #pragma unroll
    for (int j = 0; j < 4; j++) {
      v8f z = {0.f, 0.f, 0.f, 0.f, 0.f, 0.f, 0.f, 0.f};
      acc[i][j] = z;
    }

  for (int k0 = 0; k0 < K; k0 += BK) {
    // --- A: 2 async 16B DMA chunks per thread ---
    #pragma unroll
    for (int q = 0; q < 2; q++) { async_ld16(ldsa[q], ga[q]); ga[q] += BK; }
    // --- B: 8 unpredicated u32 loads -> K-pair packed via b16 stores ---
    #pragma unroll
    for (int t = 0; t < 8; t++) {
      u32 v = *(const u32*)(gb + (size_t)(4 * t) * ldb);
      int o = bs0 + t * (4 * BS_STRIDE);
      Bs16[o]     = (u16)(v & 0xFFFFu);
      Bs16[o + 2] = (u16)(v >> 16);
    }
    gb += (size_t)BK * ldb;
    if (k0 + BK < K) __builtin_prefetch(gb, 0, 0);   // global_prefetch_b8, next tile

    wait_async0();
    __syncthreads();

    union FragA { v4u q[2]; v16bf v; };
    union FragB { u32 u[8]; v16bf v; };
    FragA af[2];
    FragB bfr[4];
    const v4u* Aq = (const v4u*)As;
    // A frag (16-bit 16x32): two 16B chunks, chunk ids ((2*lhalf)^f) and ((2*lhalf+1)^f)
    #pragma unroll
    for (int i = 0; i < 2; i++) {
      int m = wrow + i * 16 + lm;
      int f = (m >> 2) & 3;
      af[i].q[0] = Aq[m * 4 + ((2 * lhalf) ^ f)];
      af[i].q[1] = Aq[m * 4 + ((2 * lhalf + 1) ^ f)];
    }
    // B frag (16-bit 32x16): lane col n=lm; VGPR v -> k-pair v + lhalf*8
    #pragma unroll
    for (int j = 0; j < 4; j++) {
      int n = wcol + j * 16 + lm;
      #pragma unroll
      for (int v = 0; v < 8; v++) {
        int kp = (lhalf << 3) + v;
        bfr[j].u[v] = Bs[kp * BS_STRIDE + n];
      }
    }
    #pragma unroll
    for (int i = 0; i < 2; i++)
      #pragma unroll
      for (int j = 0; j < 4; j++)
        acc[i][j] = __builtin_amdgcn_wmma_f32_16x16x32_bf16(
            false, af[i].v, false, bfr[j].v, (short)0, acc[i][j], false, false);
    __syncthreads();
  }

  // epilogue: C layout VGPR g -> row = g + lhalf*8, col = lm
  #pragma unroll
  for (int i = 0; i < 2; i++) {
    #pragma unroll
    for (int j = 0; j < 4; j++) {
      int col = n0 + wcol + j * 16 + lm;
      #pragma unroll
      for (int g = 0; g < 8; g++) {
        int row = m0 + wrow + i * 16 + g + lhalf * 8;
        if (row < M && col < N) {
          float val = acc[i][j][g];
          if (bias)     val += bias[col];
          if (residual) val += residual[(long)row * ldc + col];
          long o = (long)row * ldc + col;
          if (Cf)  Cf[o]  = val;
          if (Cbf) Cbf[o] = f2bf(val);
        }
      }
    }
  }
}

// ------------------------------- launcher ----------------------------------

extern "C" void kernel_launch(void* const* d_in, const int* in_sizes, int n_in,
                              void* d_out, int out_size, void* d_ws, size_t ws_size,
                              hipStream_t stream) {
  (void)in_sizes; (void)n_in; (void)out_size; (void)ws_size;

  const float* x        = (const float*)d_in[0];
  const float* m2v_x    = (const float*)d_in[1];
  const int*   label_y  = (const int*)d_in[2];
  const int*   label_idx= (const int*)d_in[3];
  const int*   edge_src = (const int*)d_in[4];
  const int*   edge_dst = (const int*)d_in[5];
  const float* m2v_W    = (const float*)d_in[6];
  const float* m2v_b    = (const float*)d_in[7];
  const float* label_emb= (const float*)d_in[8];
  const float* lmlp_W1  = (const float*)d_in[9];
  const float* lmlp_b1  = (const float*)d_in[10];
  const float* lmlp_g1  = (const float*)d_in[11];
  const float* lmlp_be1 = (const float*)d_in[12];
  const float* lmlp_W2  = (const float*)d_in[13];
  const float* lmlp_b2  = (const float*)d_in[14];
  const float* gat_fc   = (const float*)d_in[15];
  const float* attn_l   = (const float*)d_in[16];
  const float* attn_r   = (const float*)d_in[17];
  // d_in[18] = gat_bias: constant column shift, cancels inside BN -> unused
  const float* gat_g    = (const float*)d_in[19];
  const float* gat_be   = (const float*)d_in[20];
  const float* skip_W   = (const float*)d_in[21];
  const float* skip_b   = (const float*)d_in[22];
  const float* skip_g   = (const float*)d_in[23];
  const float* skip_be  = (const float*)d_in[24];
  const float* pa_W     = (const float*)d_in[25];
  const float* pa_b     = (const float*)d_in[26];
  const float* pn_g     = (const float*)d_in[27];
  const float* pn_be    = (const float*)d_in[28];
  const float* mlp_W1   = (const float*)d_in[29];
  const float* mlp_b1   = (const float*)d_in[30];
  const float* mlp_g    = (const float*)d_in[31];
  const float* mlp_be   = (const float*)d_in[32];
  const float* mlp_W2   = (const float*)d_in[33];
  const float* mlp_b2   = (const float*)d_in[34];
  float* out = (float*)d_out;

  // ---- workspace carve-up (phase-overlaid big region R1) ----
  char* ws = (char*)d_ws;
  size_t off = 0;
  auto alloc = [&](size_t bytes) -> char* {
    char* p = ws + off;
    off += (bytes + 255) & ~(size_t)255;
    return p;
  };
  const size_t MB = 1u << 20;

  u16*   x1    = (u16*)  alloc((size_t)NSRC * CC * 2);        // updated features, bf16
  char*  R1    =         alloc(208 * MB);                     // phase-overlaid scratch
  float* hconv = (float*)alloc((size_t)NE * NDST * HIDN * 4); // elu(bn(rst)) per relation
  float* sk    = (float*)alloc((size_t)NDST * HIDN * 4);
  float* ebuf  = (float*)alloc((size_t)EDGES * HEADS * 4);
  float* el    = (float*)alloc((size_t)NSRC * HEADS * 4);
  float* er    = (float*)alloc((size_t)NDST * HEADS * 4);
  float* mseg  = (float*)alloc((size_t)NDST * HEADS * 4);
  float* dseg  = (float*)alloc((size_t)NDST * HEADS * 4);
  float* mean  = (float*)alloc(1536 * 4);
  float* rstd  = (float*)alloc(1536 * 4);
  u16* w_m2v  = (u16*)alloc((size_t)64 * CC * 2);
  u16* w_l1   = (u16*)alloc((size_t)2 * CC * HIDN * 2);
  u16* w_l2   = (u16*)alloc((size_t)HIDN * CC * 2);
  u16* w_skip = (u16*)alloc((size_t)CC * HIDN * 2);
  u16* w_gat  = (u16*)alloc((size_t)NE * CC * HIDN * 2);
  u16* w_m1   = (u16*)alloc((size_t)HIDN * HIDN * 2);
  u16* w_m2   = (u16*)alloc((size_t)HIDN * NCLSP * 2);        // padded to 256 cols

  // R1 sub-views (disjoint lifetimes)
  u16*   m2vx_bf  = (u16*)  R1;               // phase A
  u16*   Lin      = (u16*)  R1;               // phase B (label MLP)
  float* hlab     = (float*)(R1 +  34 * MB);
  u16*   hlab_bf  = (u16*)  (R1 +  56 * MB);
  float* h2       = (float*)(R1 +  72 * MB);
  float* fs       = (float*)R1;               // phase C (per-relation GEMM out)
  float* fused    = (float*)R1;               // phase D (head)
  u16*   fused_bf = (u16*)  (R1 +  52 * MB);
  float* mh       = (float*)(R1 +  80 * MB);
  u16*   mh_bf    = (u16*)  (R1 + 132 * MB);
  float* logits   = (float*)(R1 + 160 * MB);

  auto cdiv = [](long a, int b) { return (int)((a + b - 1) / b); };
  auto cvt  = [&](const float* s, u16* d, long n) {
    k_f32_to_bf16<<<cdiv(n, 256), 256, 0, stream>>>(s, d, n);
  };
  auto gemm = [&](const u16* A, const u16* B, const float* bias, const float* res,
                  float* Cf, u16* Cbf, int M, int N, int K, int lda, int ldb, int ldc) {
    dim3 g((N + BN - 1) / BN, (M + BM - 1) / BM);
    k_gemm_bf16<<<g, 256, 0, stream>>>(A, B, bias, res, Cf, Cbf, M, N, K, lda, ldb, ldc);
  };

  // weight / activation conversions to bf16
  cvt(m2v_W,  w_m2v, (long)64 * CC);
  cvt(lmlp_W1, w_l1, (long)2 * CC * HIDN);
  cvt(lmlp_W2, w_l2, (long)HIDN * CC);
  cvt(skip_W, w_skip, (long)CC * HIDN);
  cvt(gat_fc, w_gat, (long)NE * CC * HIDN);
  cvt(mlp_W1, w_m1, (long)HIDN * HIDN);
  k_cvt_pad<<<cdiv((long)HIDN * NCLSP, 256), 256, 0, stream>>>(mlp_W2, w_m2, HIDN, NCLS, NCLSP);
  cvt(m2v_x, m2vx_bf, (long)NSRC * 64);

  // 1) x1 = bf16( x + m2v_x @ m2v_W + m2v_b )
  gemm(m2vx_bf, w_m2v, m2v_b, x, nullptr, x1, NSRC, CC, 64, 64, CC, CC);

  // 2) label MLP, scatter into x1
  k_label_gather<<<cdiv((long)LBL * 2 * CC, 256), 256, 0, stream>>>(label_emb, label_y, x1, label_idx, Lin, (long)LBL * 2 * CC);
  gemm(Lin, w_l1, lmlp_b1, nullptr, hlab, nullptr, LBL, HIDN, 2 * CC, 2 * CC, HIDN, HIDN);
  k_colstats<<<HIDN, 256, 0, stream>>>(hlab, LBL, HIDN, mean, rstd);
  k_bn_act<<<cdiv((long)LBL * HIDN, 256), 256, 0, stream>>>(hlab, lmlp_g1, lmlp_be1, mean, rstd, nullptr, hlab_bf, (long)LBL * HIDN, HIDN, 1);
  gemm(hlab_bf, w_l2, lmlp_b2, nullptr, h2, nullptr, LBL, CC, HIDN, HIDN, CC, CC);
  k_label_scatter<<<cdiv((long)LBL * CC, 256), 256, 0, stream>>>(h2, label_idx, x1, (long)LBL * CC);

  // 3) skip branch: elu(bn(x_dst @ skip_W + skip_b))
  gemm(x1, w_skip, skip_b, nullptr, sk, nullptr, NDST, HIDN, CC, CC, HIDN, HIDN);
  k_colstats<<<HIDN, 256, 0, stream>>>(sk, NDST, HIDN, mean, rstd);
  k_bn_act<<<cdiv((long)NDST * HIDN, 256), 256, 0, stream>>>(sk, skip_g, skip_be, mean, rstd, sk, nullptr, (long)NDST * HIDN, HIDN, 2);

  // 4) per-relation GAT
  for (int r = 0; r < NE; r++) {
    float* rst = hconv + (size_t)r * NDST * HIDN;
    gemm(x1, w_gat + (size_t)r * CC * HIDN, nullptr, nullptr, fs, nullptr, NSRC, HIDN, CC, CC, HIDN, HIDN);
    k_dot_heads<<<cdiv((long)NSRC * HEADS, 256), 256, 0, stream>>>(fs, attn_l + (size_t)r * HEADS * DHD, el, NSRC);
    k_dot_heads<<<cdiv((long)NDST * HEADS, 256), 256, 0, stream>>>(fs, attn_r + (size_t)r * HEADS * DHD, er, NDST);
    k_fill<<<cdiv((long)NDST * HEADS, 256), 256, 0, stream>>>(mseg, -__builtin_inff(), (long)NDST * HEADS);
    k_fill<<<cdiv((long)NDST * HEADS, 256), 256, 0, stream>>>(dseg, 0.f, (long)NDST * HEADS);
    k_fill<<<cdiv((long)NDST * HIDN, 256), 256, 0, stream>>>(rst, 0.f, (long)NDST * HIDN);
    const int* es = edge_src + (size_t)r * EDGES;
    const int* ed = edge_dst + (size_t)r * EDGES;
    k_edge_max<<<cdiv((long)EDGES * HEADS, 256), 256, 0, stream>>>(es, ed, el, er, ebuf, mseg, EDGES);
    k_fix_nonfinite<<<cdiv((long)NDST * HEADS, 256), 256, 0, stream>>>(mseg, (long)NDST * HEADS);
    k_edge_exp<<<cdiv((long)EDGES * HEADS, 256), 256, 0, stream>>>(ed, ebuf, mseg, dseg, EDGES);
    k_edge_aggr<<<EDGES, DHD, 0, stream>>>(es, ed, ebuf, dseg, fs, rst, EDGES);
    // gat_bias cancels inside BN; apply bn + elu in place
    k_colstats<<<HIDN, 256, 0, stream>>>(rst, NDST, HIDN, mean, rstd);
    k_bn_act<<<cdiv((long)NDST * HIDN, 256), 256, 0, stream>>>(rst, gat_g + (size_t)r * HIDN, gat_be + (size_t)r * HIDN, mean, rstd, rst, nullptr, (long)NDST * HIDN, HIDN, 2);
  }

  // 5) attention pooling over {conv0..2, sk} then bn
  k_pool<<<NDST, DHD, 0, stream>>>(hconv, sk, pa_W, pa_b, fused, NDST);
  k_colstats<<<HIDN, 256, 0, stream>>>(fused, NDST, HIDN, mean, rstd);
  k_bn_act<<<cdiv((long)NDST * HIDN, 256), 256, 0, stream>>>(fused, pn_g, pn_be, mean, rstd, nullptr, fused_bf, (long)NDST * HIDN, HIDN, 0);

  // 6) final MLP + softmax
  gemm(fused_bf, w_m1, mlp_b1, nullptr, mh, nullptr, NDST, HIDN, HIDN, HIDN, HIDN, HIDN);
  k_colstats<<<HIDN, 256, 0, stream>>>(mh, NDST, HIDN, mean, rstd);
  k_bn_act<<<cdiv((long)NDST * HIDN, 256), 256, 0, stream>>>(mh, mlp_g, mlp_be, mean, rstd, nullptr, mh_bf, (long)NDST * HIDN, HIDN, 1);
  gemm(mh_bf, w_m2, mlp_b2, nullptr, logits, nullptr, NDST, NCLS, HIDN, HIDN, NCLSP, NCLS);
  k_softmax<<<NDST, 256, 0, stream>>>(logits, out, NDST, NCLS);
}
